// Graph_Layer_44787918963014
// MI455X (gfx1250) — matrix operand order repaired and checked by reading, not compiled, not run on hardware
//
#include <hip/hip_runtime.h>
#include <hip/hip_bf16.h>
#include <math.h>

// Problem constants (match reference)
#define NROWS 8192
#define INSZ  512
#define FEATD 128
#define NOUTD 512

typedef __attribute__((ext_vector_type(16))) __bf16 v16bf;
typedef __attribute__((ext_vector_type(8)))  float  v8f;
typedef unsigned int v4u __attribute__((ext_vector_type(4)));
typedef int          v8i __attribute__((ext_vector_type(8)));
typedef int          v4i __attribute__((ext_vector_type(4)));

union BfVec {
  v16bf v;
  uint4 u[2];
};

// ---------------------------------------------------------------------------
// WMMA operand loaders (wave32, 16-bit operands; A and B share one contiguous k order)
// A (16x32 MxK): lane m=lane&15, half=lane>>4 -> one contiguous 32B chunk (k=16*half+s).
// B (32x16 KxN): lane n=lane&15 -> one contiguous 32B chunk when storage is
//   N-major (row index = N, K contiguous).
// C/D (16x16 f32): VGPR v -> row v + 8*(lane>>4), col lane&15.
// ---------------------------------------------------------------------------
__device__ __forceinline__ v16bf load_a16(const __bf16* base, int ld, int row0, int k0) {
  int lane = threadIdx.x & 31;
  int m    = lane & 15;
  int half = lane >> 4;
  const __bf16* p = base + (size_t)(row0 + m) * ld + k0 + half * 16;
  BfVec r;
  r.u[0] = *(const uint4*)(p);
  r.u[1] = *(const uint4*)(p + 8);
  return r.v;
}

__device__ __forceinline__ v16bf load_b16(const __bf16* base, int ld, int n0, int k0) {
  int lane = threadIdx.x & 31;
  int n    = lane & 15;
  int koff = (lane >> 4) << 4;
  const __bf16* p = base + (size_t)(n0 + n) * ld + k0 + koff;
  BfVec r;
  r.u[0] = *(const uint4*)(p);
  r.u[1] = *(const uint4*)(p + 8);
  return r.v;
}

__device__ __forceinline__ v8f wmma_bf16(v16bf a, v16bf b, v8f c) {
  return __builtin_amdgcn_wmma_f32_16x16x32_bf16(false, a, false, b, (short)0, c,
                                                 false, false);
}

__device__ __forceinline__ v8f zero_v8f() {
  v8f z;
  for (int i = 0; i < 8; ++i) z[i] = 0.f;
  return z;
}

// ---------------------------------------------------------------------------
// Tensor Data Mover: async 2-D tile load, global -> LDS (bf16 elements).
// Builds D# group0/group1 per CDNA5 ISA 8.3/8.4; groups 2/3 zero (2-D tile).
// This toolchain's builtin form (6 args):
//   (uint32x4 g0, int32x8 g1, int32x4 g2, int32x4 g3, int32x8 g4, i32 cpol)
// ---------------------------------------------------------------------------
__device__ __forceinline__ void tdm_load_2d(unsigned lds_addr, const void* gptr,
                                            unsigned tile_d0, unsigned tile_d1,
                                            unsigned tensor_d0, unsigned tensor_d1,
                                            unsigned long long stride0) {
  unsigned long long ga = (unsigned long long)(uintptr_t)gptr;
  v4u g0;
  g0[0] = 1u;                                   // count=1, user descriptor
  g0[1] = lds_addr;                             // lds_addr [63:32]
  g0[2] = (unsigned)(ga & 0xffffffffu);         // global_addr[31:0]
  g0[3] = (unsigned)((ga >> 32) & 0x01ffffffu)  // global_addr[56:32]
        | (2u << 30);                           // type=2 ("image")
  v8i g1;
  g1[0] = (int)(1u << 16);                      // data_size=1 (2 bytes)
  g1[1] = (int)((tensor_d0 & 0xffffu) << 16);   // tensor_dim0[15:0] @ bits 63:48
  g1[2] = (int)((tensor_d0 >> 16) | ((tensor_d1 & 0xffffu) << 16));
  g1[3] = (int)((tensor_d1 >> 16) | (tile_d0 << 16));
  g1[4] = (int)(tile_d1 & 0xffffu);             // tile_dim1 (tile_dim2=0)
  g1[5] = (int)(unsigned)(stride0 & 0xffffffffu);      // dim0 stride[31:0]
  g1[6] = (int)((unsigned)(stride0 >> 32) & 0xffffu);  // dim0 stride[47:32]
  g1[7] = 0;
  v4i g2 = {0, 0, 0, 0};
  v4i g3 = {0, 0, 0, 0};
  v8i g4 = {0, 0, 0, 0, 0, 0, 0, 0};
  __builtin_amdgcn_tensor_load_to_lds(g0, g1, g2, g3, g4, 0);
}

// ---------------------------------------------------------------------------
// Prep: x (f32) -> x_bf (row-major bf16) and x^T (bf16 [col][row])
// ---------------------------------------------------------------------------
__global__ void convert_x_kernel(const float* __restrict__ x,
                                 __bf16* __restrict__ x_bf,
                                 __bf16* __restrict__ x_bfT) {
  int t = blockIdx.x * blockDim.x + threadIdx.x;   // 0 .. NROWS*INSZ-1
  int i = t >> 9;
  int c = t & 511;
  __bf16 b = (__bf16)x[t];
  x_bf[t] = b;
  x_bfT[(size_t)c * NROWS + i] = b;
}

// weight/weight_time ([K][N] f32) -> bf16 column-major [N][K];
// W0/W1 ([FEAT][K] f32, already N-major) -> bf16 elementwise
__global__ void convert_w_kernel(const float* __restrict__ w,
                                 const float* __restrict__ wt,
                                 const float* __restrict__ W0,
                                 const float* __restrict__ W1,
                                 __bf16* __restrict__ w_cm,
                                 __bf16* __restrict__ wt_cm,
                                 __bf16* __restrict__ w0_bf,
                                 __bf16* __restrict__ w1_bf) {
  int t = blockIdx.x * blockDim.x + threadIdx.x;   // 0 .. INSZ*NOUTD-1
  int k = t >> 9;
  int n = t & 511;
  w_cm [(size_t)n * INSZ + k] = (__bf16)w[t];
  wt_cm[(size_t)n * INSZ + k] = (__bf16)wt[t];
  if (t < FEATD * INSZ) {
    w0_bf[t] = (__bf16)W0[t];
    w1_bf[t] = (__bf16)W1[t];
  }
}

// ---------------------------------------------------------------------------
// q = x @ W0^T, k = x @ W1^T via WMMA.  Wave w owns feature tile [16w,16w+16)
// for both q and k; K = 512 (16 chunks).
// ---------------------------------------------------------------------------
__global__ __launch_bounds__(256)
void qk_wmma_kernel(const __bf16* __restrict__ x_bf,
                    const __bf16* __restrict__ w0_bf,
                    const __bf16* __restrict__ w1_bf,
                    __bf16* __restrict__ q_bf,
                    __bf16* __restrict__ k_bf) {
  const int lane  = threadIdx.x & 31;
  const int w     = threadIdx.x >> 5;
  const int row0  = blockIdx.x * 16;
  const int rbase = (lane >> 4) * 8;
  const int n16   = lane & 15;

  v8f aq = zero_v8f(), ak = zero_v8f();
  for (int kc = 0; kc < INSZ / 32; ++kc) {
    v16bf a = load_a16(x_bf, INSZ, row0, kc * 32);
    aq = wmma_bf16(a, load_b16(w0_bf, INSZ, w * 16, kc * 32), aq);
    ak = wmma_bf16(a, load_b16(w1_bf, INSZ, w * 16, kc * 32), ak);
  }
  int col = w * 16 + n16;
  for (int v = 0; v < 8; ++v) {
    int row = row0 + v + rbase;
    q_bf[(size_t)row * FEATD + col] = (__bf16)aq[v];
    k_bf[(size_t)row * FEATD + col] = (__bf16)ak[v];
  }
}

// ---------------------------------------------------------------------------
// Time-affinity path via closed-form prefix sums (O(N*C) instead of O(N^2*C)).
// ---------------------------------------------------------------------------
__global__ void time_kernel(const float* __restrict__ x,
                            __bf16* __restrict__ y_bf) {
  int c = blockIdx.x * blockDim.x + threadIdx.x;
  if (c >= INSZ) return;
  double P = 0.0, Q = 0.0;
  for (int i = 0; i < NROWS; ++i) {
    double xv = (double)x[(size_t)i * INSZ + c];
    P += xv;
    Q += xv * (double)i;
  }
  const double Ptot = P, Qtot = Q;
  P = 0.0; Q = 0.0;
  for (int i = 0; i < NROWS; ++i) {
    double xv = (double)x[(size_t)i * INSZ + c];
    P += xv;
    Q += xv * (double)i;
    double A = 2.0 * (double)i * P - 2.0 * Q + Qtot - (double)i * Ptot;
    long long r = (long long)NROWS * NROWS
                - (long long)i * (i + 1) / 2
                - (long long)(NROWS - 1 - i) * (NROWS - i) / 2;
    double y = ((double)NROWS * Ptot - A) / (double)r;
    y_bf[(size_t)i * INSZ + c] = (__bf16)(float)y;
  }
}

// ---------------------------------------------------------------------------
// Flash attention, BM=32 query rows per WG, BN=64 keys per block,
// double-buffered LDS tiles filled by the Tensor Data Mover (wave 0 issues,
// TENSORcnt tracks completion, workgroup barrier publishes).
//   S:    wave w -> rowgroup w>>2, key tile (w&3)*16   (4 WMMAs)
//   P@x:  wave w -> columns [64w,64w+64), both rowgroups (16 WMMAs)
// LDS: 2*(16KB K + 64KB xT) + softmax state  (~175KB of 320KB)
// ---------------------------------------------------------------------------
#define BN 64
#define NBLK (NROWS / BN)

__global__ __launch_bounds__(256)
void attn_kernel(const __bf16* __restrict__ q_bf,
                 const __bf16* __restrict__ k_bf,
                 const __bf16* __restrict__ x_bfT,
                 __bf16* __restrict__ gx_bf) {
  __shared__ __align__(16) __bf16 lds_k[2][BN * FEATD];   // [key][feat] 2x16KB
  __shared__ __align__(16) __bf16 lds_xT[2][INSZ * BN];   // [col][key]  2x64KB
  __shared__ __align__(16) float  lds_S[32 * 68];         // padded rows
  __shared__ __align__(16) __bf16 lds_P[32 * 72];         // padded rows
  __shared__ float lds_m[32], lds_l[32], lds_scale[32];
  __shared__ float lds_red[32 * 8];

  const int tid   = threadIdx.x;
  const int lane  = tid & 31;
  const int w     = tid >> 5;             // wave 0..7
  const int row0  = blockIdx.x * 32;
  const int rg_s  = w >> 2;               // S-phase rowgroup for this wave
  const int kt_s  = w & 3;                // S-phase key tile for this wave
  const int r     = tid & 31;             // softmax row for this thread
  const int seg   = tid >> 5;             // softmax key segment (8 keys)
  const int rbase = (lane >> 4) * 8;
  const int n16   = lane & 15;

  const unsigned ldsK0 = (unsigned)(uintptr_t)(void*)&lds_k[0][0];
  const unsigned ldsK1 = (unsigned)(uintptr_t)(void*)&lds_k[1][0];
  const unsigned ldsX0 = (unsigned)(uintptr_t)(void*)&lds_xT[0][0];
  const unsigned ldsX1 = (unsigned)(uintptr_t)(void*)&lds_xT[1][0];

  // Resident q A-operands: 2 rowgroups x 4 K-chunks
  v16bf qa[2][4];
  for (int rg = 0; rg < 2; ++rg)
    for (int kc = 0; kc < 4; ++kc)
      qa[rg][kc] = load_a16(q_bf, FEATD, row0 + rg * 16, kc * 32);

  v8f acc[2][4];
  for (int rg = 0; rg < 2; ++rg)
    for (int nt = 0; nt < 4; ++nt) acc[rg][nt] = zero_v8f();

  if (tid < 32) { lds_m[tid] = -INFINITY; lds_l[tid] = 0.f; }

  // Prologue: TDM-fill buffer 0 with key block 0
  if (w == 0) {
    tdm_load_2d(ldsK0, k_bf, FEATD, BN, FEATD, NROWS, FEATD);
    tdm_load_2d(ldsX0, x_bfT, BN, INSZ, NROWS, NROWS, NROWS);
  }

  for (int kb = 0; kb < NBLK; ++kb) {
    const int p = kb & 1;
    __syncthreads();   // all waves done reading the buffer we are about to refill
    if (w == 0) {
      if (kb + 1 < NBLK) {
        const int pn = (kb + 1) & 1;
        tdm_load_2d(pn ? ldsK1 : ldsK0,
                    k_bf + (size_t)(kb + 1) * BN * FEATD,
                    FEATD, BN, FEATD, NROWS, FEATD);
        tdm_load_2d(pn ? ldsX1 : ldsX0,
                    x_bfT + (size_t)(kb + 1) * BN,
                    BN, INSZ, NROWS, NROWS, NROWS);
        __builtin_amdgcn_s_wait_tensorcnt(2);  // block kb's two loads done
      } else {
        __builtin_amdgcn_s_wait_tensorcnt(0);  // final block: drain everything
      }
    }
    __syncthreads();   // block kb's tiles are in LDS for everyone

    // ---- S tile: rowgroup rg_s x keys [16*kt_s, 16*kt_s+16) ----
    {
      v8f s = zero_v8f();
      for (int kc = 0; kc < 4; ++kc)
        s = wmma_bf16(qa[rg_s][kc],
                      load_b16(&lds_k[p][0], FEATD, kt_s * 16, kc * 32), s);
      int col = kt_s * 16 + n16;
      for (int v = 0; v < 8; ++v)
        lds_S[(rg_s * 16 + rbase + v) * 68 + col] = s[v];
    }
    __syncthreads();

    // ---- online softmax over the 64 new keys (32 rows) ----
    float lmax = -INFINITY;
    for (int j = 0; j < 8; ++j)
      lmax = fmaxf(lmax, lds_S[r * 68 + seg * 8 + j]);
    lds_red[r * 8 + seg] = lmax;
    __syncthreads();
    if (tid < 32) {
      float mb = lds_red[tid * 8];
      for (int s2 = 1; s2 < 8; ++s2) mb = fmaxf(mb, lds_red[tid * 8 + s2]);
      float mold = lds_m[tid];
      float mnew = fmaxf(mold, mb);
      lds_m[tid]     = mnew;
      lds_scale[tid] = __expf(mold - mnew);   // 0 when mold == -inf
    }
    __syncthreads();
    {
      float mnew = lds_m[r];
      float psum = 0.f;
      for (int j = 0; j < 8; ++j) {
        int key = seg * 8 + j;
        float pv = __expf(lds_S[r * 68 + key] - mnew);
        lds_P[r * 72 + key] = (__bf16)pv;
        psum += pv;
      }
      lds_red[r * 8 + seg] = psum;
    }
    __syncthreads();
    if (tid < 32) {
      float ls = 0.f;
      for (int s2 = 0; s2 < 8; ++s2) ls += lds_red[tid * 8 + s2];
      lds_l[tid] = lds_l[tid] * lds_scale[tid] + ls;
    }
    __syncthreads();

    // ---- rescale O, accumulate P @ x over this wave's 64 columns ----
    float sc0[8], sc1[8];
    for (int v = 0; v < 8; ++v) {
      sc0[v] = lds_scale[rbase + v];
      sc1[v] = lds_scale[16 + rbase + v];
    }
    for (int nt = 0; nt < 4; ++nt)
      for (int v = 0; v < 8; ++v) {
        acc[0][nt][v] *= sc0[v];
        acc[1][nt][v] *= sc1[v];
      }
    for (int kc = 0; kc < 2; ++kc) {
      v16bf pa0 = load_a16(lds_P, 72, 0,  kc * 32);
      v16bf pa1 = load_a16(lds_P, 72, 16, kc * 32);
      for (int nt = 0; nt < 4; ++nt) {
        v16bf b = load_b16(&lds_xT[p][0], BN, w * 64 + nt * 16, kc * 32);
        acc[0][nt] = wmma_bf16(pa0, b, acc[0][nt]);
        acc[1][nt] = wmma_bf16(pa1, b, acc[1][nt]);
      }
    }
  }
  __syncthreads();

  // ---- normalize and emit Gx (bf16 row-major) ----
  float invl0[8], invl1[8];
  for (int v = 0; v < 8; ++v) {
    invl0[v] = 1.f / lds_l[rbase + v];
    invl1[v] = 1.f / lds_l[16 + rbase + v];
  }
  for (int rg = 0; rg < 2; ++rg) {
    for (int nt = 0; nt < 4; ++nt) {
      int col = w * 64 + nt * 16 + n16;
      for (int v = 0; v < 8; ++v) {
        int row = row0 + rg * 16 + rbase + v;
        float nv = acc[rg][nt][v] * (rg ? invl1[v] : invl0[v]);
        gx_bf[(size_t)row * INSZ + col] = (__bf16)nv;
      }
    }
  }
}

// ---------------------------------------------------------------------------
// out = 0.5 * (Gx @ weight) + 0.5 * (y_time @ weight_time)
// ---------------------------------------------------------------------------
__global__ __launch_bounds__(256)
void out_kernel(const __bf16* __restrict__ gx_bf,
                const __bf16* __restrict__ y_bf,
                const __bf16* __restrict__ w_cm,
                const __bf16* __restrict__ wt_cm,
                float* __restrict__ out) {
  const int lane  = threadIdx.x & 31;
  const int w     = threadIdx.x >> 5;
  const int row0  = blockIdx.x * 16;
  const int col0  = w * 64;
  const int rbase = (lane >> 4) * 8;
  const int n16   = lane & 15;

  v8f aG[4], aT[4];
  for (int nt = 0; nt < 4; ++nt) { aG[nt] = zero_v8f(); aT[nt] = zero_v8f(); }

  for (int kc = 0; kc < INSZ / 32; ++kc) {
    v16bf ag = load_a16(gx_bf, INSZ, row0, kc * 32);
    v16bf at = load_a16(y_bf,  INSZ, row0, kc * 32);
    for (int nt = 0; nt < 4; ++nt) {
      aG[nt] = wmma_bf16(ag, load_b16(w_cm,  INSZ, col0 + nt * 16, kc * 32), aG[nt]);
      aT[nt] = wmma_bf16(at, load_b16(wt_cm, INSZ, col0 + nt * 16, kc * 32), aT[nt]);
    }
  }

  for (int nt = 0; nt < 4; ++nt) {
    int col = col0 + nt * 16 + n16;
    for (int v = 0; v < 8; ++v)
      out[(size_t)(row0 + v + rbase) * NOUTD + col] =
          0.5f * (aG[nt][v] + aT[nt][v]);
  }
}

// ---------------------------------------------------------------------------
// Launch
// ---------------------------------------------------------------------------
extern "C" void kernel_launch(void* const* d_in, const int* in_sizes, int n_in,
                              void* d_out, int out_size, void* d_ws, size_t ws_size,
                              hipStream_t stream) {
  const float* x           = (const float*)d_in[0];
  const float* W0          = (const float*)d_in[1];
  const float* W1          = (const float*)d_in[2];
  const float* weight      = (const float*)d_in[3];
  const float* weight_time = (const float*)d_in[4];
  float* out = (float*)d_out;

  char* ws = (char*)d_ws;
  const size_t MB = 1u << 20;
  __bf16* x_bf  = (__bf16*)(ws);                      //  8 MiB [row][col]
  __bf16* x_bfT = (__bf16*)(ws +  8 * MB);            //  8 MiB [col][row]
  __bf16* q_bf  = (__bf16*)(ws + 16 * MB);            //  2 MiB [row][feat]
  __bf16* k_bf  = (__bf16*)(ws + 18 * MB);            //  2 MiB [row][feat]
  __bf16* gx_bf = (__bf16*)(ws + 20 * MB);            //  8 MiB [row][col]
  __bf16* y_bf  = (__bf16*)(ws + 28 * MB);            //  8 MiB [row][col]
  __bf16* w_cm  = (__bf16*)(ws + 36 * MB);            // 0.5 MiB [n][k]
  __bf16* wt_cm = (__bf16*)(ws + 36 * MB + 512 * 1024);
  __bf16* w0_bf = (__bf16*)(ws + 37 * MB);            // 128 KiB [feat][k]
  __bf16* w1_bf = (__bf16*)(ws + 37 * MB + 128 * 1024);

  convert_x_kernel<<<(NROWS * INSZ) / 256, 256, 0, stream>>>(x, x_bf, x_bfT);
  convert_w_kernel<<<(INSZ * NOUTD) / 256, 256, 0, stream>>>(
      weight, weight_time, W0, W1, w_cm, wt_cm, w0_bf, w1_bf);
  qk_wmma_kernel<<<NROWS / 16, 256, 0, stream>>>(x_bf, w0_bf, w1_bf, q_bf, k_bf);
  time_kernel<<<2, 256, 0, stream>>>(x, y_bf);
  attn_kernel<<<NROWS / 32, 256, 0, stream>>>(q_bf, k_bf, x_bfT, gx_bf);
  out_kernel<<<NROWS / 16, 256, 0, stream>>>(gx_bf, y_bf, w_cm, wt_cm, out);
}
